// MultiHeadAttention_15899968929958
// MI455X (gfx1250) — matrix operand order repaired and checked by reading, not compiled, not run on hardware
//
#include <hip/hip_runtime.h>

typedef __bf16 bf16_t;
typedef __attribute__((ext_vector_type(16))) __bf16 v16bf;
typedef __attribute__((ext_vector_type(8)))  __bf16 v8bf;
typedef __attribute__((ext_vector_type(8)))  float  v8f;

#define WMMA_BF16(a, b, c) \
  __builtin_amdgcn_wmma_f32_16x16x32_bf16(false, (a), false, (b), (short)0, (c), false, false)

namespace cfg {
constexpr int B    = 2;
constexpr int S    = 2048;
constexpr int E    = 1024;
constexpr int H    = 16;
constexpr int DH   = 64;
constexpr int HALF = 128;   // window/2
constexpr int E3   = 3 * E;
}

// ---------------------------------------------------------------- helpers ---

__device__ __forceinline__ bf16_t f2bf(float f) {
  unsigned u = __builtin_bit_cast(unsigned, f);
  u += 0x7FFFu + ((u >> 16) & 1u);               // round-to-nearest-even
  unsigned short h = (unsigned short)(u >> 16);
  return __builtin_bit_cast(bf16_t, h);
}

// A-fragment: 16x32 bf16, row-major source with stride lda (elements).
// Lane l<16 : row l,   K elems {0..7, 16..23}
// Lane l>=16: row l-16, K elems {8..15, 24..31}
__device__ __forceinline__ v16bf load_frag_a(const bf16_t* base, int lda, int lane) {
  const int row = lane & 15;
  const int k0  = (lane < 16) ? 0 : 8;
  const bf16_t* p = base + (size_t)row * lda + k0;
  v8bf lo = *(const v8bf*)(p);
  v8bf hi = *(const v8bf*)(p + 16);
  v16bf r;
#pragma unroll
  for (int i = 0; i < 8; ++i) { r[i] = lo[i]; r[i + 8] = hi[i]; }
  return r;
}

// B-fragment: logical B[k][n] (32x16) where column n is a contiguous
// 32-element run in memory at rows + n*ldr. Lane l<16 holds K {0..7, 16..23}
// of column l; lane l>=16 holds K {8..15, 24..31} of column l-16.
__device__ __forceinline__ v16bf load_frag_b(const bf16_t* rows, int ldr, int lane) {
  const int n  = lane & 15;
  const int k0 = (lane < 16) ? 0 : 8;
  const bf16_t* p = rows + (size_t)n * ldr + k0;
  v8bf lo = *(const v8bf*)(p);
  v8bf hi = *(const v8bf*)(p + 16);
  v16bf r;
#pragma unroll
  for (int i = 0; i < 8; ++i) { r[i] = lo[i]; r[i + 8] = hi[i]; }
  return r;
}

__device__ __forceinline__ float half_max(float v) {
#pragma unroll
  for (int m = 8; m >= 1; m >>= 1) v = fmaxf(v, __shfl_xor(v, m, 32));
  return v;
}
__device__ __forceinline__ float half_sum(float v) {
#pragma unroll
  for (int m = 8; m >= 1; m >>= 1) v += __shfl_xor(v, m, 32);
  return v;
}

// ----------------------------------------------------------- fp32 -> bf16 ---

__global__ void cvt_bf16_kernel(const float* __restrict__ in,
                                bf16_t* __restrict__ out, int n) {
  int i = blockIdx.x * blockDim.x + threadIdx.x;
  if (i < n) out[i] = f2bf(in[i]);
}

// --------------------------------------------------------------- QKV GEMM ---
// C[m,n] = sum_k X[m,k] * Wqkv[n,k] + bqkv[n]; scatter into Q/K/V^T layouts.
// Block: 256 threads = 8 waves (2 M x 4 N); wave tile 32x64; block tile 64x256.

__global__ __launch_bounds__(256)
void gemm_qkv_kernel(const bf16_t* __restrict__ X, const bf16_t* __restrict__ Wq,
                     const float* __restrict__ bias,
                     bf16_t* __restrict__ Qm, bf16_t* __restrict__ Km,
                     bf16_t* __restrict__ Vt) {
  const int lane = threadIdx.x & 31;
  const int wave = threadIdx.x >> 5;
  const int m0 = blockIdx.x * 64 + (wave & 1) * 32;
  const int n0 = blockIdx.y * 256 + (wave >> 1) * 64;
  const int K = cfg::E;

  v8f acc[2][4] = {};
  for (int k = 0; k < K; k += 32) {
    __builtin_prefetch(X + (size_t)m0 * K + k + 256, 0, 1);
    v16bf a0 = load_frag_a(X + (size_t)(m0)      * K + k, K, lane);
    v16bf a1 = load_frag_a(X + (size_t)(m0 + 16) * K + k, K, lane);
#pragma unroll
    for (int nj = 0; nj < 4; ++nj) {
      v16bf bw = load_frag_b(Wq + (size_t)(n0 + nj * 16) * K + k, K, lane);
      acc[0][nj] = WMMA_BF16(a0, bw, acc[0][nj]);
      acc[1][nj] = WMMA_BF16(a1, bw, acc[1][nj]);
    }
  }

#pragma unroll
  for (int mi = 0; mi < 2; ++mi)
#pragma unroll
    for (int nj = 0; nj < 4; ++nj)
#pragma unroll
      for (int i = 0; i < 8; ++i) {
        int m = m0 + mi * 16 + i + ((lane < 16) ? 0 : 8);
        int n = n0 + nj * 16 + (lane & 15);
        bf16_t bv = f2bf(acc[mi][nj][i] + bias[n]);
        int bb = m / cfg::S, ss = m % cfg::S;
        int hh = n / (3 * cfg::DH), cc = n % (3 * cfg::DH);
        size_t bh = (size_t)bb * cfg::H + hh;
        if (cc < cfg::DH)
          Qm[(bh * cfg::S + ss) * cfg::DH + cc] = bv;
        else if (cc < 2 * cfg::DH)
          Km[(bh * cfg::S + ss) * cfg::DH + (cc - cfg::DH)] = bv;
        else
          Vt[(bh * cfg::DH + (cc - 2 * cfg::DH)) * cfg::S + ss] = bv;
      }
}

// -------------------------------------------------- banded flash attention ---
// One wave per 16-query tile per (b,h). Keys stepped 32 at a time.

__global__ __launch_bounds__(32)
void attn_kernel(const bf16_t* __restrict__ Qm, const bf16_t* __restrict__ Km,
                 const bf16_t* __restrict__ Vt, const int* __restrict__ pmask,
                 bf16_t* __restrict__ AO) {
  const int lane = threadIdx.x & 31;
  const int q0 = blockIdx.x * 16;
  const int h  = blockIdx.y;
  const int b  = blockIdx.z;
  const size_t bh = (size_t)b * cfg::H + h;
  const bf16_t* Qbh = Qm + (bh * cfg::S + q0) * cfg::DH;
  const bf16_t* Kbh = Km + bh * cfg::S * cfg::DH;
  const bf16_t* Vbh = Vt + bh * cfg::DH * cfg::S;
  const int* pm = pmask + (size_t)b * cfg::S;

  const int qhi  = (lane < 16) ? 0 : 8;   // C-layout row offset for this half
  const int col  = lane & 15;
  const int kofs = (lane < 16) ? 0 : 16;
  const float LOG2E = 1.44269504f;

  v16bf aq0 = load_frag_a(Qbh, cfg::DH, lane);       // d = 0..31
  v16bf aq1 = load_frag_a(Qbh + 32, cfg::DH, lane);  // d = 32..63

  v8f o[4] = {};
  float mrow[8], srow[8];
#pragma unroll
  for (int i = 0; i < 8; ++i) { mrow[i] = -3.0e38f; srow[i] = 0.0f; }

  int jlo = q0 - cfg::HALF; if (jlo < 0) jlo = 0; jlo &= ~31;
  int jhi = q0 + 15 + cfg::HALF; if (jhi > cfg::S - 1) jhi = cfg::S - 1;

  __shared__ __align__(64) bf16_t Pld[16 * 32];

  for (int j = jlo; j <= jhi; j += 32) {
    // scores for 32 keys: two 16x16 tiles
    v16bf bk00 = load_frag_b(Kbh + (size_t)(j)      * cfg::DH + 0,  cfg::DH, lane);
    v16bf bk01 = load_frag_b(Kbh + (size_t)(j)      * cfg::DH + 32, cfg::DH, lane);
    v16bf bk10 = load_frag_b(Kbh + (size_t)(j + 16) * cfg::DH + 0,  cfg::DH, lane);
    v16bf bk11 = load_frag_b(Kbh + (size_t)(j + 16) * cfg::DH + 32, cfg::DH, lane);
    v8f s0 = {}, s1 = {};
    s0 = WMMA_BF16(aq0, bk00, s0); s0 = WMMA_BF16(aq1, bk01, s0);
    s1 = WMMA_BF16(aq0, bk10, s1); s1 = WMMA_BF16(aq1, bk11, s1);

    const int k_lo = j + col, k_hi = j + 16 + col;
    const bool pad_lo = (pm[k_lo] != 0), pad_hi = (pm[k_hi] != 0);

    float alpha[8];
#pragma unroll
    for (int i = 0; i < 8; ++i) {
      int q = q0 + i + qhi;
      int d0 = q - k_lo; if (d0 < 0) d0 = -d0;
      int d1 = q - k_hi; if (d1 < 0) d1 = -d1;
      bool ok0 = pad_lo && (d0 <= cfg::HALF);
      bool ok1 = pad_hi && (d1 <= cfg::HALF);
      float v0 = ok0 ? s0[i] * 0.125f : -3.0e38f;  // 1/sqrt(64)
      float v1 = ok1 ? s1[i] * 0.125f : -3.0e38f;
      float nm = fmaxf(mrow[i], half_max(fmaxf(v0, v1)));
      float al = __builtin_exp2f((mrow[i] - nm) * LOG2E);
      float e0 = ok0 ? __builtin_exp2f((v0 - nm) * LOG2E) : 0.0f;
      float e1 = ok1 ? __builtin_exp2f((v1 - nm) * LOG2E) : 0.0f;
      srow[i] = srow[i] * al + half_sum(e0 + e1);
      mrow[i] = nm;
      alpha[i] = al;
      Pld[(i + qhi) * 32 + col]      = f2bf(e0);
      Pld[(i + qhi) * 32 + 16 + col] = f2bf(e1);
    }
#pragma unroll
    for (int t = 0; t < 4; ++t)
#pragma unroll
      for (int i = 0; i < 8; ++i) o[t][i] *= alpha[i];

    __syncthreads();
    v16bf pa = load_frag_a(Pld, 32, lane);  // P: 16 queries x 32 keys
#pragma unroll
    for (int t = 0; t < 4; ++t) {
      v16bf bv = load_frag_b(Vbh + (size_t)(t * 16) * cfg::S + j, cfg::S, lane);
      o[t] = WMMA_BF16(pa, bv, o[t]);
    }
    __syncthreads();
  }

  float inv[8];
#pragma unroll
  for (int i = 0; i < 8; ++i) {
    int q = q0 + i + qhi;
    inv[i] = (srow[i] > 0.0f && pm[q] != 0) ? 1.0f / srow[i] : 0.0f;
  }
#pragma unroll
  for (int t = 0; t < 4; ++t)
#pragma unroll
    for (int i = 0; i < 8; ++i) {
      int q = q0 + i + qhi;
      AO[((size_t)b * cfg::S + q) * cfg::E + h * cfg::DH + t * 16 + col] =
          f2bf(o[t][i] * inv[i]);
    }
}

// ------------------------------------------------------------ output GEMM ---
// out[m,n] = sum_k AO[m,k] * Wo[n,k] + bo[n]  (fp32 store)

__global__ __launch_bounds__(256)
void gemm_out_kernel(const bf16_t* __restrict__ A, const bf16_t* __restrict__ Wo,
                     const float* __restrict__ bias, float* __restrict__ out) {
  const int lane = threadIdx.x & 31;
  const int wave = threadIdx.x >> 5;
  const int m0 = blockIdx.x * 64 + (wave & 1) * 32;
  const int n0 = blockIdx.y * 256 + (wave >> 1) * 64;
  const int K = cfg::E;

  v8f acc[2][4] = {};
  for (int k = 0; k < K; k += 32) {
    __builtin_prefetch(A + (size_t)m0 * K + k + 256, 0, 1);
    v16bf a0 = load_frag_a(A + (size_t)(m0)      * K + k, K, lane);
    v16bf a1 = load_frag_a(A + (size_t)(m0 + 16) * K + k, K, lane);
#pragma unroll
    for (int nj = 0; nj < 4; ++nj) {
      v16bf bw = load_frag_b(Wo + (size_t)(n0 + nj * 16) * K + k, K, lane);
      acc[0][nj] = WMMA_BF16(a0, bw, acc[0][nj]);
      acc[1][nj] = WMMA_BF16(a1, bw, acc[1][nj]);
    }
  }
#pragma unroll
  for (int mi = 0; mi < 2; ++mi)
#pragma unroll
    for (int nj = 0; nj < 4; ++nj)
#pragma unroll
      for (int i = 0; i < 8; ++i) {
        int m = m0 + mi * 16 + i + ((lane < 16) ? 0 : 8);
        int n = n0 + nj * 16 + (lane & 15);
        out[(size_t)m * cfg::E + n] = acc[mi][nj][i] + bias[n];
      }
}

// ------------------------------------------------------------------ launch ---

extern "C" void kernel_launch(void* const* d_in, const int* in_sizes, int n_in,
                              void* d_out, int out_size, void* d_ws, size_t ws_size,
                              hipStream_t stream) {
  (void)in_sizes; (void)n_in; (void)out_size; (void)ws_size;
  using namespace cfg;

  const float* x     = (const float*)d_in[0];
  const int*   pmask = (const int*)d_in[1];
  const float* Wqkv  = (const float*)d_in[2];
  const float* bqkv  = (const float*)d_in[3];
  const float* Wo    = (const float*)d_in[4];
  const float* bo    = (const float*)d_in[5];
  float* out = (float*)d_out;

  char* ws = (char*)d_ws;
  constexpr size_t SZ_X16 = (size_t)B * S * E * 2;       // 8 MiB
  constexpr size_t SZ_WQ  = (size_t)3 * E * E * 2;       // 6 MiB
  constexpr size_t SZ_WO  = (size_t)E * E * 2;           // 2 MiB
  constexpr size_t SZ_HB  = (size_t)B * H * S * DH * 2;  // 8 MiB each
  bf16_t* X16 = (bf16_t*)(ws);
  bf16_t* WQ  = (bf16_t*)(ws + SZ_X16);
  bf16_t* WOb = (bf16_t*)(ws + SZ_X16 + SZ_WQ);
  bf16_t* Qm  = (bf16_t*)(ws + SZ_X16 + SZ_WQ + SZ_WO);
  bf16_t* Km  = (bf16_t*)(ws + SZ_X16 + SZ_WQ + SZ_WO + SZ_HB);
  bf16_t* Vt  = (bf16_t*)(ws + SZ_X16 + SZ_WQ + SZ_WO + 2 * SZ_HB);
  bf16_t* AO  = (bf16_t*)(ws + SZ_X16 + SZ_WQ + SZ_WO + 3 * SZ_HB);

  const int NX = B * S * E, NWQ = 3 * E * E, NWO = E * E;
  cvt_bf16_kernel<<<dim3((NX  + 255) / 256), 256, 0, stream>>>(x,    X16, NX);
  cvt_bf16_kernel<<<dim3((NWQ + 255) / 256), 256, 0, stream>>>(Wqkv, WQ,  NWQ);
  cvt_bf16_kernel<<<dim3((NWO + 255) / 256), 256, 0, stream>>>(Wo,   WOb, NWO);

  gemm_qkv_kernel<<<dim3((B * S) / 64, E3 / 256), 256, 0, stream>>>(
      X16, WQ, bqkv, Qm, Km, Vt);

  attn_kernel<<<dim3(S / 16, H, B), 32, 0, stream>>>(Qm, Km, Vt, pmask, AO);

  gemm_out_kernel<<<dim3((B * S) / 64, E / 256), 256, 0, stream>>>(
      AO, WOb, bo, out);
}
